// PolicyNetwork_76613626626306
// MI455X (gfx1250) — matrix operand order repaired and checked by reading, not compiled, not run on hardware
//
#include <hip/hip_runtime.h>
#include <hip/hip_bf16.h>
#include <math.h>

typedef __attribute__((ext_vector_type(16))) __bf16 v16bf;
typedef __attribute__((ext_vector_type(8)))  float  v8f;
typedef __attribute__((ext_vector_type(4)))  unsigned int u32x4;
typedef unsigned short u16;
typedef unsigned int   u32;

// ---- workspace layout (per net: 64 KB; net stride in u16 units) ----
#define NETWS_BYTES 65536
#define NETWS_U16   (NETWS_BYTES / 2)
#define FREG_U16    16384            // f32 bias/head region starts at byte 32768
// f32 region offsets (in floats)
#define OFF_BHID 0                   // b0..b3 : 4*64
#define OFF_B4   256                 // b4 padded to 16
#define OFF_AW0  272                 // aW0 : up to 7*16
#define OFF_AB0  384                 // ab0 : 16
#define OFF_AW1  400                 // aW1 : 16*8
#define OFF_AB1  528                 // ab1 : 8
#define OFF_AW2  536                 // aW2 : 8
#define OFF_AB2  544                 // ab2 : 1

__device__ __forceinline__ u16 f2bf(float x) {
    u32 u = __float_as_uint(x);
    u32 r = u + 0x7FFFu + ((u >> 16) & 1u);   // round-to-nearest-even
    return (u16)(r >> 16);
}

union Frag {
    v16bf bf;
    u16   h[16];
    u32x4 q[2];
};

struct AllParams { const float* p[32]; };  // 16 pos leaves then 16 rot leaves

// =====================================================================
// Kernel 1: pre-swizzle weights into WMMA B-fragment layout (bf16)
//   B-frag (32x16 bf16, contiguous k order, same as A): lane L holds column
//   n = n0*16 + (L&15); halves h=0..15 hold K = kc + (L>>4)*16 + h.
//   Fragment order per net: L0: n0=0..3 (4) | L1..L3: kc{0,32} x n0 (8 ea) | L4: kc{0,32} (2)
// =====================================================================
__global__ __launch_bounds__(256) void prep_weights(AllParams ap, u16* __restrict__ wsu)
{
    const int tid = threadIdx.x;
    for (int net = 0; net < 2; ++net) {
        const float* const* P = ap.p + net * 16;
        const int in0  = net ? 23 : 21;
        const int out4 = net ? 6 : 3;
        u16*   fragbase = wsu + net * NETWS_U16;
        float* fr = (float*)(wsu + net * NETWS_U16 + FREG_U16);

        // ---- matrix fragments: 30 frags * 32 lanes = 960 units ----
        for (int u = tid; u < 960; u += 256) {
            const int f = u >> 5, lane = u & 31;
            int layer, kc, n0;
            if (f < 4)       { layer = 0; kc = 0;                n0 = f;      }
            else if (f < 12) { int g = f - 4;  layer = 1; kc = (g >> 2) * 32; n0 = g & 3; }
            else if (f < 20) { int g = f - 12; layer = 2; kc = (g >> 2) * 32; n0 = g & 3; }
            else if (f < 28) { int g = f - 20; layer = 3; kc = (g >> 2) * 32; n0 = g & 3; }
            else             { layer = 4; kc = (f - 28) * 32;    n0 = 0;      }
            const float* Wp = P[layer * 2];                 // W0,W1,W2,W3,W4
            const int Kin = (layer == 0) ? in0 : 64;        // valid K rows
            const int Nt  = (layer == 4) ? out4 : 64;       // valid N cols (also row stride)
            const int n   = n0 * 16 + (lane & 15);
            const int kb  = kc + (lane >> 4) * 16;
            u16* dst = fragbase + u * 16;
            #pragma unroll
            for (int h = 0; h < 16; ++h) {
                const int k = kb + h;
                const float v = (k < Kin && n < Nt) ? Wp[k * Nt + n] : 0.0f;
                dst[h] = f2bf(v);
            }
        }

        // ---- biases + attention head params (f32) ----
        for (int i = tid; i < 64; i += 256) {
            fr[OFF_BHID +   0 + i] = P[1][i];   // b0
            fr[OFF_BHID +  64 + i] = P[3][i];   // b1
            fr[OFF_BHID + 128 + i] = P[5][i];   // b2
            fr[OFF_BHID + 192 + i] = P[7][i];   // b3
        }
        for (int i = tid; i < 16;  i += 256) fr[OFF_B4  + i] = (i < out4) ? P[9][i] : 0.0f;
        for (int i = tid; i < 112; i += 256) fr[OFF_AW0 + i] = (i < (out4 + 1) * 16) ? P[10][i] : 0.0f;
        for (int i = tid; i < 16;  i += 256) fr[OFF_AB0 + i] = P[11][i];
        for (int i = tid; i < 128; i += 256) fr[OFF_AW1 + i] = P[12][i];
        for (int i = tid; i < 8;   i += 256) fr[OFF_AB1 + i] = P[13][i];
        for (int i = tid; i < 8;   i += 256) fr[OFF_AW2 + i] = P[14][i];
        if (tid == 0) fr[OFF_AB2] = P[15][0];
    }
}

// =====================================================================
// Kernel 2: fully fused network. 1 workgroup per batch item b.
//   8 waves: wave w -> net = w>>2 (0 pos / 1 rot), row-tile = w&3 (16 rows).
//   Activations live in a private 16x64 bf16 LDS slice per wave.
// =====================================================================
__global__ __launch_bounds__(256) void policy_main(
    const float* __restrict__ pos_edges,  const float* __restrict__ pos_dists,
    const float* __restrict__ input_poses, const float* __restrict__ rot_offset,
    const float* __restrict__ rot_pref,   const float* __restrict__ rot_dists,
    const u16* __restrict__ wsu, float* __restrict__ out)
{
    __shared__ u16   actS[8 * 16 * 64];   // 16 KB: per-wave 16x64 bf16 activations
    __shared__ float outS[2 * 64 * 16];   //  8 KB: final-layer outputs (padded to 16)
    __shared__ float logitS[128];         // attention logits (2 nets x 64)

    const int tid  = threadIdx.x;
    const int lane = tid & 31;
    const int w    = tid >> 5;
    const int b    = blockIdx.x;
    const int net  = w >> 2;
    const int tile = w & 3;
    const int rowbase = b * 64 + tile * 16;

    const u16*   frag = wsu + net * NETWS_U16;
    const float* fr   = (const float*)(wsu + net * NETWS_U16 + FREG_U16);
    u16* myact = &actS[w * 16 * 64];

    // ---------------- layer-0 features -> LDS (bf16, K padded to 32) ----------------
    if (lane < 16) {
        const int gi = rowbase + lane;
        u16* dst = &myact[lane * 64];
        if (net == 0) {
            const float* e = pos_edges + (long)gi * 21;
            #pragma unroll
            for (int j = 0; j < 21; ++j) dst[j] = f2bf(e[j]);
            #pragma unroll
            for (int j = 21; j < 32; ++j) dst[j] = 0;
        } else {
            const float* ip = input_poses + (long)gi * 9;
            const float* ro = rot_offset  + (long)gi * 4;
            const float Rx0 = ip[3], Rx1 = ip[4], Rx2 = ip[5];
            const float Ry0 = ip[6], Ry1 = ip[7], Ry2 = ip[8];
            const float Rz0 = Rx1 * Ry2 - Rx2 * Ry1;
            const float Rz1 = Rx2 * Ry0 - Rx0 * Ry2;
            const float Rz2 = Rx0 * Ry1 - Rx1 * Ry0;
            const float qx = ro[0], qy = ro[1], qz = ro[2], qw = ro[3];
            const float inv = rsqrtf(qx * qx + qy * qy + qz * qz + qw * qw);
            const float q1 = qx * inv, q2 = qy * inv, q3 = qz * inv, q0 = qw * inv;
            const float r00 = 2.f * (q0 * q0 + q1 * q1) - 1.f;
            const float r10 = 2.f * (q1 * q2 + q0 * q3);
            const float r20 = 2.f * (q1 * q3 - q0 * q2);
            const float r01 = 2.f * (q1 * q2 - q0 * q3);
            const float r11 = 2.f * (q0 * q0 + q2 * q2) - 1.f;
            const float r21 = 2.f * (q2 * q3 + q0 * q1);
            dst[0] = f2bf(Rx0 * r00 + Ry0 * r10 + Rz0 * r20);
            dst[1] = f2bf(Rx1 * r00 + Ry1 * r10 + Rz1 * r20);
            dst[2] = f2bf(Rx2 * r00 + Ry2 * r10 + Rz2 * r20);
            dst[3] = f2bf(Rx0 * r01 + Ry0 * r11 + Rz0 * r21);
            dst[4] = f2bf(Rx1 * r01 + Ry1 * r11 + Rz1 * r21);
            dst[5] = f2bf(Rx2 * r01 + Ry2 * r11 + Rz2 * r21);
            const float* pf = rot_pref + (long)gi * 16;
            #pragma unroll
            for (int j = 0; j < 16; ++j) dst[6 + j] = f2bf(pf[j]);
            dst[22] = f2bf(rot_dists[gi]);
            #pragma unroll
            for (int j = 23; j < 32; ++j) dst[j] = 0;
        }
    }
    __syncthreads();

    // A layout (16-bit 16x32): lane row = lane&15; half = lane>>4;
    //   halves 0-7 : K = kc + half*16 + j       (contiguous 16 B)
    //   halves 8-15: K = kc + half*16 + 8 + j   (contiguous 16 B)
    const int dn   = lane & 15;          // also the C/D column within a tile
    const int half = lane >> 4;
    const int dm   = half * 8;           // C/D row base
    const u16* arow = &myact[dn * 64];

    Frag A0, A1;
    v8f acc[4];

    // ---------------- layer 0: K=32, N=64 ----------------
    A0.q[0] = *(const u32x4*)&arow[half * 16];
    A0.q[1] = *(const u32x4*)&arow[8 + half * 16];
    #pragma unroll
    for (int n0 = 0; n0 < 4; ++n0) {
        Frag B0;
        const u16* bp = &frag[(n0 * 32 + lane) * 16];
        B0.q[0] = *(const u32x4*)&bp[0];
        B0.q[1] = *(const u32x4*)&bp[8];
        v8f c = {};
        c = __builtin_amdgcn_wmma_f32_16x16x32_bf16(false, A0.bf, false, B0.bf,
                                                    (short)0, c, false, false);
        acc[n0] = c;
    }
    {
        const float* bias = fr + OFF_BHID;
        #pragma unroll
        for (int n0 = 0; n0 < 4; ++n0)
            #pragma unroll
            for (int i = 0; i < 8; ++i) {
                float v = acc[n0][i] + bias[n0 * 16 + dn];
                myact[(dm + i) * 64 + n0 * 16 + dn] = f2bf(fmaxf(v, 0.f));
            }
    }

    // ---------------- layers 1..3: K=64, N=64 ----------------
    #pragma unroll 1
    for (int l = 1; l < 4; ++l) {
        __syncthreads();
        A0.q[0] = *(const u32x4*)&arow[half * 16];
        A0.q[1] = *(const u32x4*)&arow[8 + half * 16];
        A1.q[0] = *(const u32x4*)&arow[32 + half * 16];
        A1.q[1] = *(const u32x4*)&arow[40 + half * 16];
        const int fb = 4 + (l - 1) * 8;
        #pragma unroll
        for (int n0 = 0; n0 < 4; ++n0) {
            Frag B0, B1;
            const u16* b0p = &frag[((fb + n0) * 32 + lane) * 16];
            const u16* b1p = &frag[((fb + 4 + n0) * 32 + lane) * 16];
            B0.q[0] = *(const u32x4*)&b0p[0];  B0.q[1] = *(const u32x4*)&b0p[8];
            B1.q[0] = *(const u32x4*)&b1p[0];  B1.q[1] = *(const u32x4*)&b1p[8];
            v8f c = {};
            c = __builtin_amdgcn_wmma_f32_16x16x32_bf16(false, A0.bf, false, B0.bf,
                                                        (short)0, c, false, false);
            c = __builtin_amdgcn_wmma_f32_16x16x32_bf16(false, A1.bf, false, B1.bf,
                                                        (short)0, c, false, false);
            acc[n0] = c;
        }
        const float* bias = fr + OFF_BHID + l * 64;
        #pragma unroll
        for (int n0 = 0; n0 < 4; ++n0)
            #pragma unroll
            for (int i = 0; i < 8; ++i) {
                float v = acc[n0][i] + bias[n0 * 16 + dn];
                myact[(dm + i) * 64 + n0 * 16 + dn] = f2bf(fmaxf(v, 0.f));
            }
    }

    // ---------------- layer 4: K=64, N=16 (padded), no ReLU ----------------
    __syncthreads();
    A0.q[0] = *(const u32x4*)&arow[half * 16];
    A0.q[1] = *(const u32x4*)&arow[8 + half * 16];
    A1.q[0] = *(const u32x4*)&arow[32 + half * 16];
    A1.q[1] = *(const u32x4*)&arow[40 + half * 16];
    {
        Frag B0, B1;
        const u16* b0p = &frag[(28 * 32 + lane) * 16];
        const u16* b1p = &frag[(29 * 32 + lane) * 16];
        B0.q[0] = *(const u32x4*)&b0p[0];  B0.q[1] = *(const u32x4*)&b0p[8];
        B1.q[0] = *(const u32x4*)&b1p[0];  B1.q[1] = *(const u32x4*)&b1p[8];
        v8f c = {};
        c = __builtin_amdgcn_wmma_f32_16x16x32_bf16(false, A0.bf, false, B0.bf,
                                                    (short)0, c, false, false);
        c = __builtin_amdgcn_wmma_f32_16x16x32_bf16(false, A1.bf, false, B1.bf,
                                                    (short)0, c, false, false);
        const float* b4 = fr + OFF_B4;
        #pragma unroll
        for (int i = 0; i < 8; ++i)
            outS[(net * 64 + tile * 16 + dm + i) * 16 + dn] = c[i] + b4[dn];
    }
    __syncthreads();

    // ---------------- attention head: one thread per (net,row) ----------------
    if (tid < 128) {
        const int hn = tid >> 6;          // net
        const int n  = tid & 63;          // row within batch item
        const int outd = hn ? 6 : 3;
        const float* hf = (const float*)(wsu + hn * NETWS_U16 + FREG_U16);
        const float* ob = &outS[(hn * 64 + n) * 16];
        float a[7];
        #pragma unroll
        for (int c2 = 0; c2 < 7; ++c2) a[c2] = 0.f;
        for (int c2 = 0; c2 < outd; ++c2) a[c2] = ob[c2];
        const float* dp = hn ? rot_dists : pos_dists;
        a[outd] = dp[b * 64 + n];

        float h0[16];
        #pragma unroll
        for (int k = 0; k < 16; ++k) {
            float s = hf[OFF_AB0 + k];
            for (int j = 0; j <= outd; ++j) s += a[j] * hf[OFF_AW0 + j * 16 + k];
            h0[k] = fmaxf(s, 0.f);
        }
        float h1[8];
        #pragma unroll
        for (int k = 0; k < 8; ++k) {
            float s = hf[OFF_AB1 + k];
            #pragma unroll
            for (int j = 0; j < 16; ++j) s += h0[j] * hf[OFF_AW1 + j * 8 + k];
            h1[k] = fmaxf(s, 0.f);
        }
        float lg = hf[OFF_AB2];
        #pragma unroll
        for (int j = 0; j < 8; ++j) lg += h1[j] * hf[OFF_AW2 + j];
        logitS[hn * 64 + n] = lg;
    }
    __syncthreads();

    // ---------------- softmax over N=64 + weighted sum: wave 0 = pos, wave 1 = rot ----------------
    if (w < 2) {
        const int rn   = w;
        const int outd = rn ? 6 : 3;
        const float l1 = logitS[rn * 64 + lane];
        const float l2 = logitS[rn * 64 + lane + 32];
        float mx = fmaxf(l1, l2);
        #pragma unroll
        for (int off = 16; off > 0; off >>= 1) mx = fmaxf(mx, __shfl_xor(mx, off, 32));
        const float e1 = __expf(l1 - mx), e2 = __expf(l2 - mx);
        float s = e1 + e2;
        #pragma unroll
        for (int off = 16; off > 0; off >>= 1) s += __shfl_xor(s, off, 32);
        const float invs = 1.f / s;
        for (int c2 = 0; c2 < outd; ++c2) {
            float v = e1 * outS[(rn * 64 + lane) * 16 + c2]
                    + e2 * outS[(rn * 64 + lane + 32) * 16 + c2];
            #pragma unroll
            for (int off = 16; off > 0; off >>= 1) v += __shfl_xor(v, off, 32);
            if (lane == 0) out[b * 9 + (rn ? 3 : 0) + c2] = v * invs;
        }
    }
}

// =====================================================================
extern "C" void kernel_launch(void* const* d_in, const int* in_sizes, int n_in,
                              void* d_out, int out_size, void* d_ws, size_t ws_size,
                              hipStream_t stream)
{
    (void)in_sizes; (void)n_in; (void)out_size; (void)ws_size;
    AllParams ap;
    for (int i = 0; i < 32; ++i) ap.p[i] = (const float*)d_in[6 + i];
    u16* wsu = (u16*)d_ws;

    prep_weights<<<1, 256, 0, stream>>>(ap, wsu);
    policy_main<<<4096, 256, 0, stream>>>(
        (const float*)d_in[0], (const float*)d_in[1], (const float*)d_in[2],
        (const float*)d_in[3], (const float*)d_in[4], (const float*)d_in[5],
        wsu, (float*)d_out);
}
